// CustomLlamaModel_18717467476139
// MI455X (gfx1250) — compile-verified, matrix operand order checked by reading, not run on hardware
//
#include <hip/hip_runtime.h>
#include <stdint.h>

#define S_LEN 2048
#define H_DIM 1024
#define HQ_N  16
#define D_HEAD 64
#define FF_DIM 4096
#define N_LAYER 2
#define EPS_RMS 1e-6f

typedef __attribute__((ext_vector_type(16))) __bf16 v16bf;
typedef __attribute__((ext_vector_type(8)))  float  v8f;

union Frag { v16bf v; uint4 q[2]; };

__device__ __forceinline__ uint16_t f2bf(float f) {
    union { float f; uint32_t u; } a; a.f = f;
    uint32_t r = a.u + 0x7FFFu + ((a.u >> 16) & 1u);   // round-to-nearest-even
    return (uint16_t)(r >> 16);
}
__device__ __forceinline__ float bf2f(uint16_t h) {
    union { uint32_t u; float f; } a; a.u = ((uint32_t)h) << 16;
    return a.f;
}

// gfx1250 async copy: global memory -> LDS, 16 bytes per lane, ASYNCcnt-tracked
__device__ __forceinline__ void async_b128(uint32_t lds_byte_off, const void* gaddr) {
    asm volatile("global_load_async_to_lds_b128 %0, %1, off"
                 :: "v"(lds_byte_off), "v"(gaddr) : "memory");
}

// ---------------- embedding gather ----------------
__global__ void k_embed(const int* __restrict__ ids, const float* __restrict__ emb,
                        float* __restrict__ x) {
    int i = blockIdx.x * blockDim.x + threadIdx.x;
    if (i >= S_LEN * H_DIM) return;
    int s = i >> 10, h = i & 1023;
    x[i] = emb[(size_t)ids[s] * H_DIM + h];
}

// ---------------- fp32 -> bf16 convert ----------------
__global__ void k_cvt(const float* __restrict__ src, uint16_t* __restrict__ dst, int n) {
    int i = blockIdx.x * blockDim.x + threadIdx.x;
    if (i < n) dst[i] = f2bf(src[i]);
}

// ---------------- RMSNorm (one block per row) ----------------
__global__ __launch_bounds__(256) void k_rmsnorm(const float* __restrict__ x,
                                                 const float* __restrict__ w,
                                                 uint16_t* __restrict__ out) {
    __shared__ float red[256];
    int row = blockIdx.x;
    const float* xr = x + (size_t)row * H_DIM;
    float ss = 0.f;
    for (int i = threadIdx.x; i < H_DIM; i += 256) { float v = xr[i]; ss += v * v; }
    red[threadIdx.x] = ss;
    __syncthreads();
    for (int st = 128; st > 0; st >>= 1) {
        if (threadIdx.x < st) red[threadIdx.x] += red[threadIdx.x + st];
        __syncthreads();
    }
    float rs = rsqrtf(red[0] * (1.f / H_DIM) + EPS_RMS);
    for (int i = threadIdx.x; i < H_DIM; i += 256)
        out[(size_t)row * H_DIM + i] = f2bf(xr[i] * rs * w[i]);
}

// ---------------- WMMA GEMM, double-buffered async-to-LDS pipeline -------------
// C[M,N] = A[M,K](bf16) @ W[N,K]^T(bf16) (+resid | bf16-out)
// Block tile 128(M) x 64(N), K-step 32. 8 wave32 waves arranged 4(m) x 2(n);
// each wave computes a 32x32 register tile (4 accumulators, 2x2 fragment reuse).
// K-step i+1 is staged via global_load_async_to_lds_b128 while step i's wmma run;
// s_wait_asynccnt 3 retires only the previous step's 3 copies (in-order ASYNCcnt).
// LDS rows padded to 40 elems (80B) => 16-lane fragment reads are bank-conflict-free.
#define LROW 40
__global__ __launch_bounds__(256) void k_gemm(const uint16_t* __restrict__ A,
                                              const uint16_t* __restrict__ W,
                                              const float* __restrict__ resid,
                                              float* __restrict__ C,
                                              uint16_t* __restrict__ Cb,
                                              int M, int N, int K) {
    __shared__ __align__(16) uint16_t sA[2][128 * LROW];
    __shared__ __align__(16) uint16_t sB[2][64 * LROW];

    const int t = threadIdx.x;
    const int wave = t >> 5, lane = t & 31;
    const int hl = lane >> 4, l16 = lane & 15;
    const int wm = wave & 3;          // 0..3 -> 32-row strip
    const int wn = wave >> 2;         // 0..1 -> 32-col strip
    const int m_blk = blockIdx.y * 128;
    const int n_blk = blockIdx.x * 64;

    // staging assignments (per K-step): A = 512 x 16B chunks, B = 256 x 16B chunks
    const int c_row = t >> 2, c_part = t & 3;
    const int c_row2 = (t + 256) >> 2;
    const uint32_t offA0 = (uint32_t)(c_row * LROW + c_part * 8) * 2;
    const uint32_t offA1 = (uint32_t)(c_row2 * LROW + c_part * 8) * 2;
    const uint32_t offB  = offA0;
    const uint32_t baseA[2] = { (uint32_t)(uintptr_t)&sA[0][0], (uint32_t)(uintptr_t)&sA[1][0] };
    const uint32_t baseB[2] = { (uint32_t)(uintptr_t)&sB[0][0], (uint32_t)(uintptr_t)&sB[1][0] };
    const uint16_t* gA0 = A + (size_t)(m_blk + c_row)  * K + c_part * 8;
    const uint16_t* gA1 = A + (size_t)(m_blk + c_row2) * K + c_part * 8;
    const uint16_t* gB  = W + (size_t)(n_blk + c_row)  * K + c_part * 8;

    auto stage = [&](int buf, int kk) {
        async_b128(baseA[buf] + offA0, gA0 + kk);
        async_b128(baseA[buf] + offA1, gA1 + kk);
        async_b128(baseB[buf] + offB,  gB  + kk);
    };

    v8f acc[4]; acc[0] = {}; acc[1] = {}; acc[2] = {}; acc[3] = {};

    const int nsteps = K >> 5;
    stage(0, 0);
    for (int i = 0; i < nsteps; ++i) {
        const int buf = i & 1;
        if (i + 1 < nsteps) {
            stage(buf ^ 1, (i + 1) << 5);
            asm volatile("s_wait_asynccnt 3" ::: "memory"); // prev 3 copies resident
        } else {
            asm volatile("s_wait_asynccnt 0" ::: "memory");
        }
        __syncthreads();

        Frag fa[2], fb[2];
#pragma unroll
        for (int mi = 0; mi < 2; ++mi) {
            const uint16_t* p = &sA[buf][(wm * 32 + mi * 16 + l16) * LROW + hl * 8];
            fa[mi].q[0] = *(const uint4*)p;
            fa[mi].q[1] = *(const uint4*)(p + 16);
        }
#pragma unroll
        for (int ni = 0; ni < 2; ++ni) {
            const uint16_t* p = &sB[buf][(wn * 32 + ni * 16 + l16) * LROW + hl * 8];
            fb[ni].q[0] = *(const uint4*)p;
            fb[ni].q[1] = *(const uint4*)(p + 16);
        }
        acc[0] = __builtin_amdgcn_wmma_f32_16x16x32_bf16(false, fa[0].v, false, fb[0].v,
                                                         (short)0, acc[0], false, false);
        acc[1] = __builtin_amdgcn_wmma_f32_16x16x32_bf16(false, fa[0].v, false, fb[1].v,
                                                         (short)0, acc[1], false, false);
        acc[2] = __builtin_amdgcn_wmma_f32_16x16x32_bf16(false, fa[1].v, false, fb[0].v,
                                                         (short)0, acc[2], false, false);
        acc[3] = __builtin_amdgcn_wmma_f32_16x16x32_bf16(false, fa[1].v, false, fb[1].v,
                                                         (short)0, acc[3], false, false);
        __syncthreads();
    }

#pragma unroll
    for (int mi = 0; mi < 2; ++mi) {
#pragma unroll
        for (int ni = 0; ni < 2; ++ni) {
            const v8f a = acc[mi * 2 + ni];
#pragma unroll
            for (int r = 0; r < 8; ++r) {
                int row = m_blk + wm * 32 + mi * 16 + hl * 8 + r;
                int col = n_blk + wn * 32 + ni * 16 + l16;
                float v = a[r];
                if (Cb) {
                    Cb[(size_t)row * N + col] = f2bf(v);
                } else {
                    if (resid) v += resid[(size_t)row * N + col];
                    C[(size_t)row * N + col] = v;
                }
            }
        }
    }
}

// ---------------- RoPE: fp32 (S, nh, 64) -> bf16, half-split rotation ----------
__global__ void k_rope(const float* __restrict__ in, uint16_t* __restrict__ out, int nh) {
    int idx = blockIdx.x * blockDim.x + threadIdx.x;
    int total = S_LEN * nh * 32;
    if (idx >= total) return;
    int i = idx & 31;
    int t = idx >> 5;
    int h = t % nh, s = t / nh;
    float inv = __expf(-((float)(2 * i) / 64.f) * 9.210340372f); // ln(10000)
    float fr = (float)s * inv;
    float c = __cosf(fr), sn = __sinf(fr);
    size_t base = (size_t)s * nh * 64 + (size_t)h * 64;
    float x1 = in[base + i], x2 = in[base + 32 + i];
    out[base + i]      = f2bf(x1 * c - x2 * sn);
    out[base + 32 + i] = f2bf(x2 * c + x1 * sn);
}

// ---------------- V transpose: (S,64) fp32 -> (64,S) bf16 ----------------
__global__ void k_vT(const float* __restrict__ v, uint16_t* __restrict__ vT) {
    int idx = blockIdx.x * blockDim.x + threadIdx.x;
    if (idx >= S_LEN * 64) return;
    int d = idx & 63, s = idx >> 6;
    vT[(size_t)d * S_LEN + s] = f2bf(v[(size_t)s * 64 + d]);
}

// ---------------- SiLU(g) * u, bf16 in/out ----------------
__global__ void k_silu(const uint16_t* __restrict__ g, const uint16_t* __restrict__ u,
                       uint16_t* __restrict__ t, int n) {
    int i = blockIdx.x * blockDim.x + threadIdx.x;
    if (i >= n) return;
    float gg = bf2f(g[i]), uu = bf2f(u[i]);
    float sg = gg / (1.f + __expf(-gg));
    t[i] = f2bf(sg * uu);
}

// ---------------- Flash attention (MQA): one wave per (head, 16-query tile) ---
__global__ __launch_bounds__(32) void k_attn(const uint16_t* __restrict__ qbf,
                                             const uint16_t* __restrict__ kbf,
                                             const uint16_t* __restrict__ vT,
                                             uint16_t* __restrict__ outbf) {
    __shared__ __align__(16) uint16_t pbuf[16 * 32];
    const int lane = threadIdx.x, hl = lane >> 4, l16 = lane & 15;
    const int q0 = blockIdx.x * 16, h = blockIdx.y;
    const float scale = 0.125f; // 1/sqrt(64)

    Frag aQ[2];
    {
        const uint16_t* qr = qbf + (size_t)(q0 + l16) * H_DIM + h * 64 + hl * 8;
        aQ[0].q[0] = *(const uint4*)(qr);
        aQ[0].q[1] = *(const uint4*)(qr + 16);
        aQ[1].q[0] = *(const uint4*)(qr + 32);
        aQ[1].q[1] = *(const uint4*)(qr + 48);
    }

    v8f o[4]; o[0] = {}; o[1] = {}; o[2] = {}; o[3] = {};
    float mrow[8], lrow[8];
#pragma unroll
    for (int r = 0; r < 8; ++r) { mrow[r] = -3.0e38f; lrow[r] = 0.f; }

    for (int kb = 0; kb < S_LEN; kb += 32) {
        v8f s0 = {}, s1 = {};
#pragma unroll
        for (int c = 0; c < 2; ++c) {
            Frag b0, b1;
            const uint16_t* kr0 = kbf + (size_t)(kb + l16) * 64 + c * 32 + hl * 8;
            const uint16_t* kr1 = kbf + (size_t)(kb + 16 + l16) * 64 + c * 32 + hl * 8;
            b0.q[0] = *(const uint4*)(kr0); b0.q[1] = *(const uint4*)(kr0 + 16);
            b1.q[0] = *(const uint4*)(kr1); b1.q[1] = *(const uint4*)(kr1 + 16);
            s0 = __builtin_amdgcn_wmma_f32_16x16x32_bf16(false, aQ[c].v, false, b0.v,
                                                         (short)0, s0, false, false);
            s1 = __builtin_amdgcn_wmma_f32_16x16x32_bf16(false, aQ[c].v, false, b1.v,
                                                         (short)0, s1, false, false);
        }
        __syncthreads(); // pbuf reuse guard
#pragma unroll
        for (int r = 0; r < 8; ++r) {
            float v0 = s0[r] * scale, v1 = s1[r] * scale;
            float mx = fmaxf(v0, v1);
            mx = fmaxf(mx, __shfl_xor(mx, 1, 32));
            mx = fmaxf(mx, __shfl_xor(mx, 2, 32));
            mx = fmaxf(mx, __shfl_xor(mx, 4, 32));
            mx = fmaxf(mx, __shfl_xor(mx, 8, 32));
            float newm = fmaxf(mrow[r], mx);
            float fr = __expf(mrow[r] - newm);
            mrow[r] = newm;
            float p0 = __expf(v0 - newm), p1 = __expf(v1 - newm);
            float sum = p0 + p1;
            sum += __shfl_xor(sum, 1, 32);
            sum += __shfl_xor(sum, 2, 32);
            sum += __shfl_xor(sum, 4, 32);
            sum += __shfl_xor(sum, 8, 32);
            lrow[r] = lrow[r] * fr + sum;
            o[0][r] *= fr; o[1][r] *= fr; o[2][r] *= fr; o[3][r] *= fr;
            int Mr = hl * 8 + r;
            pbuf[Mr * 32 + l16]      = f2bf(p0);
            pbuf[Mr * 32 + 16 + l16] = f2bf(p1);
        }
        __syncthreads();
        Frag aP;
        const uint16_t* pr = &pbuf[l16 * 32 + hl * 8];
        aP.q[0] = *(const uint4*)(pr);
        aP.q[1] = *(const uint4*)(pr + 16);
#pragma unroll
        for (int t = 0; t < 4; ++t) {
            Frag bv;
            const uint16_t* vr = vT + (size_t)(t * 16 + l16) * S_LEN + kb + hl * 8;
            bv.q[0] = *(const uint4*)(vr);
            bv.q[1] = *(const uint4*)(vr + 16);
            o[t] = __builtin_amdgcn_wmma_f32_16x16x32_bf16(false, aP.v, false, bv.v,
                                                           (short)0, o[t], false, false);
        }
    }
#pragma unroll
    for (int t = 0; t < 4; ++t) {
#pragma unroll
        for (int r = 0; r < 8; ++r) {
            int row = q0 + hl * 8 + r;
            outbf[(size_t)row * H_DIM + h * 64 + t * 16 + l16] = f2bf(o[t][r] / lrow[r]);
        }
    }
}

// ======================= host orchestration =======================
extern "C" void kernel_launch(void* const* d_in, const int* in_sizes, int n_in,
                              void* d_out, int out_size, void* d_ws, size_t ws_size,
                              hipStream_t stream) {
    const int*   ids   = (const int*)d_in[0];
    const float* emb   = (const float*)d_in[1];
    const float* Wq    = (const float*)d_in[2];
    const float* Wk    = (const float*)d_in[3];
    const float* Wv    = (const float*)d_in[4];
    const float* Wo    = (const float*)d_in[5];
    const float* Wg    = (const float*)d_in[6];
    const float* Wu    = (const float*)d_in[7];
    const float* Wd    = (const float*)d_in[8];
    const float* ln1   = (const float*)d_in[9];
    const float* ln2   = (const float*)d_in[10];

    uintptr_t ws = (uintptr_t)d_ws;
    auto alloc = [&](size_t bytes) -> void* {
        void* p = (void*)ws;
        ws += (bytes + 255) & ~(size_t)255;
        return p;
    };
    float*    x      = (float*)alloc((size_t)S_LEN * H_DIM * 4);
    uint16_t* hbf    = (uint16_t*)alloc((size_t)S_LEN * H_DIM * 2);
    float*    qf     = (float*)alloc((size_t)S_LEN * H_DIM * 4);
    float*    kf     = (float*)alloc((size_t)S_LEN * 64 * 4);
    float*    vf     = (float*)alloc((size_t)S_LEN * 64 * 4);
    uint16_t* qbf    = (uint16_t*)alloc((size_t)S_LEN * H_DIM * 2);
    uint16_t* kbf    = (uint16_t*)alloc((size_t)S_LEN * 64 * 2);
    uint16_t* vTbf   = (uint16_t*)alloc((size_t)64 * S_LEN * 2);
    uint16_t* attnbf = (uint16_t*)alloc((size_t)S_LEN * H_DIM * 2);
    uint16_t* gbf    = (uint16_t*)alloc((size_t)S_LEN * FF_DIM * 2);
    uint16_t* ubf    = (uint16_t*)alloc((size_t)S_LEN * FF_DIM * 2);
    uint16_t* tbf    = (uint16_t*)alloc((size_t)S_LEN * FF_DIM * 2);
    uint16_t* wq_bf  = (uint16_t*)alloc((size_t)H_DIM * H_DIM * 2);
    uint16_t* wk_bf  = (uint16_t*)alloc((size_t)64 * H_DIM * 2);
    uint16_t* wv_bf  = (uint16_t*)alloc((size_t)64 * H_DIM * 2);
    uint16_t* wo_bf  = (uint16_t*)alloc((size_t)H_DIM * H_DIM * 2);
    uint16_t* wg_bf  = (uint16_t*)alloc((size_t)FF_DIM * H_DIM * 2);
    uint16_t* wu_bf  = (uint16_t*)alloc((size_t)FF_DIM * H_DIM * 2);
    uint16_t* wd_bf  = (uint16_t*)alloc((size_t)H_DIM * FF_DIM * 2);

    auto cvt = [&](const float* src, uint16_t* dst, int n) {
        k_cvt<<<(n + 255) / 256, 256, 0, stream>>>(src, dst, n);
    };
    auto gemm = [&](const uint16_t* A, const uint16_t* W, const float* resid,
                    float* C, uint16_t* Cb, int M, int N, int K) {
        dim3 g(N / 64, M / 128);
        k_gemm<<<g, 256, 0, stream>>>(A, W, resid, C, Cb, M, N, K);
    };

    k_embed<<<(S_LEN * H_DIM + 255) / 256, 256, 0, stream>>>(ids, emb, x);

    for (int l = 0; l < N_LAYER; ++l) {
        cvt(Wq + (size_t)l * H_DIM * H_DIM, wq_bf, H_DIM * H_DIM);
        cvt(Wk + (size_t)l * 64 * H_DIM, wk_bf, 64 * H_DIM);
        cvt(Wv + (size_t)l * 64 * H_DIM, wv_bf, 64 * H_DIM);
        cvt(Wo + (size_t)l * H_DIM * H_DIM, wo_bf, H_DIM * H_DIM);
        cvt(Wg + (size_t)l * FF_DIM * H_DIM, wg_bf, FF_DIM * H_DIM);
        cvt(Wu + (size_t)l * FF_DIM * H_DIM, wu_bf, FF_DIM * H_DIM);
        cvt(Wd + (size_t)l * H_DIM * FF_DIM, wd_bf, H_DIM * FF_DIM);

        // attention block
        k_rmsnorm<<<S_LEN, 256, 0, stream>>>(x, ln1 + (size_t)l * H_DIM, hbf);
        gemm(hbf, wq_bf, nullptr, qf, nullptr, S_LEN, H_DIM, H_DIM);
        gemm(hbf, wk_bf, nullptr, kf, nullptr, S_LEN, 64, H_DIM);
        gemm(hbf, wv_bf, nullptr, vf, nullptr, S_LEN, 64, H_DIM);
        k_rope<<<(S_LEN * HQ_N * 32 + 255) / 256, 256, 0, stream>>>(qf, qbf, HQ_N);
        k_rope<<<(S_LEN * 1 * 32 + 255) / 256, 256, 0, stream>>>(kf, kbf, 1);
        k_vT<<<(S_LEN * 64 + 255) / 256, 256, 0, stream>>>(vf, vTbf);
        k_attn<<<dim3(S_LEN / 16, HQ_N), 32, 0, stream>>>(qbf, kbf, vTbf, attnbf);
        gemm(attnbf, wo_bf, x, x, nullptr, S_LEN, H_DIM, H_DIM); // x += attn @ Wo^T

        // MLP block
        k_rmsnorm<<<S_LEN, 256, 0, stream>>>(x, ln2 + (size_t)l * H_DIM, hbf);
        gemm(hbf, wg_bf, nullptr, nullptr, gbf, S_LEN, FF_DIM, H_DIM);
        gemm(hbf, wu_bf, nullptr, nullptr, ubf, S_LEN, FF_DIM, H_DIM);
        k_silu<<<(S_LEN * FF_DIM + 255) / 256, 256, 0, stream>>>(gbf, ubf, tbf,
                                                                 S_LEN * FF_DIM);
        float* outp = (l == N_LAYER - 1) ? (float*)d_out : x;
        gemm(tbf, wd_bf, x, outp, nullptr, S_LEN, H_DIM, FF_DIM); // out = x + t @ Wd^T
    }
}
